// E2V_layer_27393301414293
// MI455X (gfx1250) — matrix operand order repaired and checked by reading, not compiled, not run on hardware
//
#include <hip/hip_runtime.h>

typedef __attribute__((ext_vector_type(16))) _Float16     v16h;
typedef __attribute__((ext_vector_type(8)))  float        v8f;
typedef __attribute__((ext_vector_type(8)))  unsigned int v8u;

#define D_E   64
#define D_IN  192
#define D_OUT 64
#define KCH   6     // 192 / 32 K-chunks
#define NTILE 4     // 64 / 16 N-tiles
#define WPB   8     // waves per block (256 threads, wave32)

__global__ __launch_bounds__(256)
void e2v_wmma_kernel(const float* __restrict__ he,   // [n_edges, 64]
                     const float* __restrict__ hn,   // [n_inc, 64]
                     const int*   __restrict__ aff,  // [2, n_inc] int32
                     const float* __restrict__ W,    // [64, 192] row-major [out,in]
                     const float* __restrict__ bias, // [64]
                     float*       __restrict__ out,  // [n_inc, 64]
                     int n_inc, int n_tiles)
{
    // B fragments of W as f16 hi/lo split, pre-laid-out in WMMA B register order.
    // sB[kc][nt][hl][lane][v] : lane holds column n = nt*16 + lane%16,
    //   u32 v packs K = kc*32 + (lane>=16 ? 16:0) + {2v (lo16), 2v+1 (hi16)}
    __shared__ unsigned int sB[KCH][NTILE][2][32][8];   // 48 KB

    for (int e = threadIdx.x; e < KCH * NTILE * 32 * 8; e += blockDim.x) {
        int v  = e & 7;
        int ln = (e >> 3) & 31;
        int nt = (e >> 8) & 3;
        int kc = e >> 10;
        int n  = nt * 16 + (ln & 15);
        int k  = kc * 32 + ((ln & 16) ? 16 : 0) + 2 * v;
        float w0 = W[n * D_IN + k];
        float w1 = W[n * D_IN + k + 1];
        _Float16 h0 = (_Float16)w0, h1 = (_Float16)w1;
        _Float16 l0 = (_Float16)(w0 - (float)h0);
        _Float16 l1 = (_Float16)(w1 - (float)h1);
        union { _Float16 h[2]; unsigned int u; } ph, pl;
        ph.h[0] = h0; ph.h[1] = h1;
        pl.h[0] = l0; pl.h[1] = l1;
        sB[kc][nt][0][ln][v] = ph.u;
        sB[kc][nt][1][ln][v] = pl.u;
    }
    __syncthreads();

    const int lane = threadIdx.x & 31;
    const int wid  = threadIdx.x >> 5;
    const int mloc = lane & 15;          // A row / output column within 16-tile
    const int hi16 = (lane >> 4) & 1;    // lane group selects K (A) / row half (C)

    float bv[NTILE];
#pragma unroll
    for (int nt = 0; nt < NTILE; ++nt) bv[nt] = bias[nt * 16 + mloc];

    for (int tile = (int)blockIdx.x * WPB + wid; tile < n_tiles;
         tile += (int)gridDim.x * WPB) {
        const int rowBase = tile * 16;
        int r = rowBase + mloc;
        if (r >= n_inc) r = n_inc - 1;   // clamp gather; stores are guarded

        const long long e0 = aff[r];
        const long long e1 = aff[(size_t)n_inc + r];
        const float* s0 = he + (size_t)e0 * D_E;     // K   0.. 63
        const float* s1 = he + (size_t)e1 * D_E;     // K  64..127
        const float* s2 = hn + (size_t)r  * D_E;     // K 128..191

        v8f acc[NTILE] = {};

#pragma unroll
        for (int kc = 0; kc < KCH; ++kc) {
            const float* base = (kc < 2) ? s0 : ((kc < 4) ? s1 : s2);
            const float* src  = base + (kc & 1) * 32 + hi16 * 16;

            // A fragment: halves 0..15 = K k0..k0+15 (k0 = 16*hi16)
            float4 f0 = *(const float4*)(src);
            float4 f1 = *(const float4*)(src + 4);
            float4 f2 = *(const float4*)(src + 8);
            float4 f3 = *(const float4*)(src + 12);
            float xs[16] = {f0.x, f0.y, f0.z, f0.w,  f1.x, f1.y, f1.z, f1.w,
                            f2.x, f2.y, f2.z, f2.w,  f3.x, f3.y, f3.z, f3.w};
            v16h ah, al;
#pragma unroll
            for (int j = 0; j < 16; ++j) {
                _Float16 h = (_Float16)xs[j];
                ah[j] = h;
                al[j] = (_Float16)(xs[j] - (float)h);   // exact residual
            }

#pragma unroll
            for (int nt = 0; nt < NTILE; ++nt) {
                v16h bh = __builtin_bit_cast(v16h, *(const v8u*)&sB[kc][nt][0][lane][0]);
                v16h bl = __builtin_bit_cast(v16h, *(const v8u*)&sB[kc][nt][1][lane][0]);
                // (xh+xl)(wh+wl) ~= xh*wh + xh*wl + xl*wh  (f32 accumulate)
                acc[nt] = __builtin_amdgcn_wmma_f32_16x16x32_f16(
                    false, ah, false, bh, (short)0, acc[nt], false, false);
                acc[nt] = __builtin_amdgcn_wmma_f32_16x16x32_f16(
                    false, ah, false, bl, (short)0, acc[nt], false, false);
                acc[nt] = __builtin_amdgcn_wmma_f32_16x16x32_f16(
                    false, al, false, bh, (short)0, acc[nt], false, false);
            }
        }

        // C/D layout: VGPR v -> row rowBase + 8*hi16 + v, col = nt*16 + mloc
        const int roff = rowBase + hi16 * 8;
#pragma unroll
        for (int nt = 0; nt < NTILE; ++nt) {
#pragma unroll
            for (int v = 0; v < 8; ++v) {
                int row = roff + v;
                if (row < n_inc) {
                    float x = acc[nt][v] + bv[nt];
                    out[(size_t)row * D_OUT + nt * 16 + mloc] = x > 0.0f ? x : 0.0f;
                }
            }
        }
    }
}

extern "C" void kernel_launch(void* const* d_in, const int* in_sizes, int n_in,
                              void* d_out, int out_size, void* d_ws, size_t ws_size,
                              hipStream_t stream) {
    const float* he   = (const float*)d_in[0];
    const float* hn   = (const float*)d_in[1];
    const int*   aff  = (const int*)d_in[2];    // int32 (JAX default x64-disabled)
    const float* W    = (const float*)d_in[3];
    const float* bias = (const float*)d_in[4];
    float*       out  = (float*)d_out;

    const int n_inc   = in_sizes[1] / D_E;      // 2,000,000
    const int n_tiles = (n_inc + 15) / 16;      // 125,000

    int blocks = (n_tiles + WPB - 1) / WPB;
    if (blocks > 2048) blocks = 2048;           // grid-stride amortizes LDS W-build

    e2v_wmma_kernel<<<blocks, 256, 0, stream>>>(he, hn, aff, W, bias, out,
                                                n_inc, n_tiles);
}
